// GRU_1365799600438
// MI455X (gfx1250) — matrix operand order repaired and checked by reading, not compiled, not run on hardware
//
#include <hip/hip_runtime.h>

typedef __bf16 bf16_t;
typedef __attribute__((ext_vector_type(8)))  __bf16 v8bf;
typedef __attribute__((ext_vector_type(16))) __bf16 v16bf;
typedef __attribute__((ext_vector_type(8)))  float  v8f;

#define Bq 128
#define Tq 512
#define Iq 256
#define Hq 512

#define HROW 520        // LDS h row stride (bf16 elems): 1040B = 260 DW, 260%64=4
#define XROW 264        // LDS/global x row stride (bf16 elems): 528B = 132 DW
#define XBLK (16 * XROW)  // 4224 elems = 8448 B per (chunk,t) x block

__device__ __forceinline__ bf16_t f2bf(float f) {
  union { float f; unsigned u; } c; c.f = f;
  unsigned u = c.u;
  u += 0x7FFFu + ((u >> 16) & 1u);           // round-to-nearest-even
  unsigned short h = (unsigned short)(u >> 16);
  bf16_t r; __builtin_memcpy(&r, &h, 2);
  return r;
}
__device__ __forceinline__ float bf2f(bf16_t b) {
  unsigned short h; __builtin_memcpy(&h, &b, 2);
  union { unsigned u; float f; } c; c.u = ((unsigned)h) << 16;
  return c.f;
}
__device__ __forceinline__ float fast_sigmoid(float x) {
  x = fminf(fmaxf(x, -30.f), 30.f);
  return 1.0f / (1.0f + __expf(-x));
}
__device__ __forceinline__ float fast_tanh(float x) {
  x = fminf(fmaxf(x, -15.f), 15.f);
  float e = __expf(2.0f * x);
  return (e - 1.0f) / (e + 1.0f);
}
__device__ __forceinline__ v16bf cat16(v8bf lo, v8bf hi) {
  return __builtin_shufflevector(lo, hi, 0,1,2,3,4,5,6,7,8,9,10,11,12,13,14,15);
}

// Issue one 16B async global->LDS copy (GLOBAL_LOAD_ASYNC_TO_LDS_B128, ASYNCcnt).
__device__ __forceinline__ void async_copy16(unsigned lds_off, const bf16_t* gsrc) {
  asm volatile("global_load_async_to_lds_b128 %0, %1, off"
               :: "v"(lds_off), "v"(gsrc) : "memory");
}
__device__ __forceinline__ void wait_async0() {
  asm volatile("s_wait_asynccnt 0x0" ::: "memory");
}
__device__ __forceinline__ unsigned lds_off_of(const void* p) {
  return (unsigned)(size_t)p;   // LDS flat addr: offset lives in addr[31:0]
}

// ---------------------------------------------------------------------------
// Pack [W_ih | W_hh] (3H x 768) into per-wave WMMA B-fragment panels, bf16.
// Layout: [j(32)][kp(24)][gate(3)][lane(32)][16] ; lane%16 = column within
// tile, element i of lane half h holds K = 8h+i (i<8) / 16+8h+(i-8) (i>=8).
// ---------------------------------------------------------------------------
__global__ void pack_gru_weights(const float* __restrict__ W_ih,
                                 const float* __restrict__ W_hh,
                                 bf16_t* __restrict__ Wpack) {
  int idx = blockIdx.x * blockDim.x + threadIdx.x;
  if (idx >= 32 * 24 * 3 * 32 * 16) return;
  int i    = idx & 15;
  int lane = (idx >> 4) & 31;
  int g    = (idx >> 9) % 3;
  int kp   = (idx / (512 * 3)) % 24;
  int j    = idx / (512 * 3 * 24);
  int c = g * Hq + j * 16 + (lane & 15);
  int k = kp * 32 + (lane >> 4) * 8 + (i & 7) + (i >> 3) * 16;
  float v = (k < Iq) ? W_ih[(size_t)c * Iq + k]
                     : W_hh[(size_t)c * Hq + (k - Iq)];
  Wpack[idx] = f2bf(v);
}

// Pack W_att (H x H) -> [ntile(32)][kp(16)][lane(32)][16] bf16 B-fragments.
__global__ void pack_att_weights(const float* __restrict__ W_att,
                                 bf16_t* __restrict__ Apack) {
  int idx = blockIdx.x * blockDim.x + threadIdx.x;
  if (idx >= 32 * 16 * 32 * 16) return;
  int i    = idx & 15;
  int lane = (idx >> 4) & 31;
  int kp   = (idx >> 9) & 15;
  int nt   = idx >> 13;
  int c = nt * 16 + (lane & 15);
  int k = kp * 32 + (lane >> 4) * 8 + (i & 7) + (i >> 3) * 16;
  Apack[idx] = f2bf(W_att[(size_t)c * Hq + k]);
}

// Pre-cast x to bf16 in per-(chunk,t) contiguous blocks matching the padded
// LDS row layout, so the recurrent kernel can async-DMA them verbatim.
__global__ void pack_x(const float* __restrict__ x, bf16_t* __restrict__ xpack) {
  int idx = blockIdx.x * blockDim.x + threadIdx.x;
  if (idx >= Bq * Tq * Iq) return;
  int k     = idx & 255;
  int row   = (idx >> 8) & 15;
  int t     = (idx >> 12) & 511;
  int chunk = idx >> 21;
  int b = chunk * 16 + row;
  xpack[(size_t)(chunk * Tq + t) * XBLK + row * XROW + k] =
      f2bf(x[((size_t)b * Tq + t) * Iq + k]);
}

// ---------------------------------------------------------------------------
// Recurrent GRU. One workgroup per 16-row batch chunk (8 WGs), 32 waves.
// Wave j owns gate columns [16j,16j+16). x_t staged by double-buffered
// GLOBAL_LOAD_ASYNC_TO_LDS; h kept in LDS (bf16) + registers (D-fragment).
// ---------------------------------------------------------------------------
__global__ __launch_bounds__(1024) void gru_recurrent(
    const bf16_t* __restrict__ xpack, const float* __restrict__ b_ih,
    const float* __restrict__ b_hh, const bf16_t* __restrict__ Wpack,
    bf16_t* __restrict__ hs) {
  __shared__ bf16_t sH[16 * HROW];     // 16,640 B
  __shared__ bf16_t sX[2][XBLK];       // 2 x 8,448 B

  const int tid   = threadIdx.x;
  const int wave  = tid >> 5;      // 0..31 = column tile j
  const int lane  = tid & 31;
  const int lrow  = lane & 15;
  const int lhalf = lane >> 4;
  const int chunk = blockIdx.x;    // 0..7

  const int   c   = wave * 16 + lrow;            // column within a gate
  const float brs = b_ih[c]          + b_hh[c];
  const float bzs = b_ih[Hq + c]     + b_hh[Hq + c];
  const float bin = b_ih[2 * Hq + c];
  const float bhn = b_hh[2 * Hq + c];

  for (int i = tid; i < 16 * HROW; i += 1024) sH[i] = f2bf(0.f);

  const bf16_t* xblk  = xpack + (size_t)chunk * Tq * XBLK;
  const unsigned xoff0 = lds_off_of(&sX[0][0]);
  const unsigned xoff1 = lds_off_of(&sX[1][0]);

  // prefetch x block for t = 0 into buffer 0  (528 x 16B = 8448 B)
  if (tid < 528) async_copy16(xoff0 + tid * 16, xblk + tid * 8);

  const bf16_t* wj = Wpack + (size_t)wave * (24 * 3 * 512) + lane * 16;

  v8f hprev = {};
  int cur = 0;

  for (int t = 0; t < Tq; ++t) {
    wait_async0();
    __syncthreads();   // x[t] resident; prev step's h writes visible

    // kick off async DMA of x[t+1] into the other buffer (overlaps GEMM)
    if (t + 1 < Tq && tid < 528)
      async_copy16((cur ? xoff0 : xoff1) + tid * 16,
                   xblk + (size_t)(t + 1) * XBLK + tid * 8);

    const bf16_t* sXc = sX[cur];

    v8f accr = {}, accz = {}, accxn = {}, acchn = {};

    // K over x (0..255): r, z, and x-part of n
    #pragma unroll 2
    for (int kp = 0; kp < 8; ++kp) {
      int ao = lrow * XROW + kp * 32 + lhalf * 8;
      v16bf a = cat16(*(const v8bf*)&sXc[ao], *(const v8bf*)&sXc[ao + 16]);
      const bf16_t* wb = wj + kp * (3 * 512);
      v16bf b0 = cat16(*(const v8bf*)(wb),        *(const v8bf*)(wb + 8));
      v16bf b1 = cat16(*(const v8bf*)(wb + 512),  *(const v8bf*)(wb + 520));
      v16bf b2 = cat16(*(const v8bf*)(wb + 1024), *(const v8bf*)(wb + 1032));
      accr  = __builtin_amdgcn_wmma_f32_16x16x32_bf16(false, a, false, b0, (short)0, accr,  false, false);
      accz  = __builtin_amdgcn_wmma_f32_16x16x32_bf16(false, a, false, b1, (short)0, accz,  false, false);
      accxn = __builtin_amdgcn_wmma_f32_16x16x32_bf16(false, a, false, b2, (short)0, accxn, false, false);
    }
    // K over h (0..511): r, z, and h-part of n  (weight panels kp 8..23)
    #pragma unroll 2
    for (int kp = 0; kp < 16; ++kp) {
      int ao = lrow * HROW + kp * 32 + lhalf * 8;
      v16bf a = cat16(*(const v8bf*)&sH[ao], *(const v8bf*)&sH[ao + 16]);
      const bf16_t* wb = wj + (8 + kp) * (3 * 512);
      v16bf b0 = cat16(*(const v8bf*)(wb),        *(const v8bf*)(wb + 8));
      v16bf b1 = cat16(*(const v8bf*)(wb + 512),  *(const v8bf*)(wb + 520));
      v16bf b2 = cat16(*(const v8bf*)(wb + 1024), *(const v8bf*)(wb + 1032));
      accr  = __builtin_amdgcn_wmma_f32_16x16x32_bf16(false, a, false, b0, (short)0, accr,  false, false);
      accz  = __builtin_amdgcn_wmma_f32_16x16x32_bf16(false, a, false, b1, (short)0, accz,  false, false);
      acchn = __builtin_amdgcn_wmma_f32_16x16x32_bf16(false, a, false, b2, (short)0, acchn, false, false);
    }
    __syncthreads();   // all waves done reading sH before overwrite

    #pragma unroll
    for (int i = 0; i < 8; ++i) {
      float r  = fast_sigmoid(accr[i] + brs);
      float z  = fast_sigmoid(accz[i] + bzs);
      float n  = fast_tanh(accxn[i] + bin + r * (acchn[i] + bhn));
      float hn = n + z * (hprev[i] - n);         // (1-z)*n + z*h
      hprev[i] = hn;
      int m = i + 8 * lhalf;
      bf16_t hb = f2bf(hn);
      sH[m * HROW + c] = hb;                     // feed next step's A
      hs[((size_t)t * Bq + chunk * 16 + m) * Hq + c] = hb;
    }
    cur ^= 1;
    // next iteration's top barrier orders these writes vs. next K-loop
  }
}

// ---------------------------------------------------------------------------
// Attention pooling with online softmax over T. Grid: (32 col tiles, 8 batch
// chunks), 16 waves. W_att tile async-DMA'd to LDS; per-wave running
// (max, denom, numer) fragments; log-sum-exp merge across waves through LDS.
// ---------------------------------------------------------------------------
__global__ __launch_bounds__(512) void attention_pool(
    const bf16_t* __restrict__ hs, const bf16_t* __restrict__ Apack,
    const float* __restrict__ b_att, float* __restrict__ out) {
  __shared__ union alignas(16) {
    bf16_t w[16 * 32 * 16];      // 16 KB W_att tile fragments
    float  r[3 * 16 * 256];      // 48 KB merge buffers (M, D, N)
  } sh;

  const int tid   = threadIdx.x;
  const int wave  = tid >> 5;    // 0..15
  const int lane  = tid & 31;
  const int lrow  = lane & 15;
  const int lhalf = lane >> 4;
  const int ntile = blockIdx.x;  // 0..31
  const int chunk = blockIdx.y;  // 0..7

  {
    const bf16_t* src = Apack + (size_t)ntile * (16 * 32 * 16);
    unsigned base = lds_off_of(&sh.w[0]);
    for (int i = tid; i < 1024; i += 512)       // 1024 x 16B = 16 KB
      async_copy16(base + i * 16, src + i * 8);
    wait_async0();
  }
  __syncthreads();

  const float batt = b_att[ntile * 16 + lrow];

  v8f M, D, N;
  #pragma unroll
  for (int i = 0; i < 8; ++i) { M[i] = -3.0e38f; D[i] = 0.f; N[i] = 0.f; }

  for (int t = wave; t < Tq; t += 16) {
    v8f acc = {};
    const bf16_t* arow = hs + ((size_t)t * Bq + chunk * 16 + lrow) * Hq;
    #pragma unroll 4
    for (int kp = 0; kp < 16; ++kp) {
      int ko = kp * 32 + lhalf * 8;
      v16bf a = cat16(*(const v8bf*)(arow + ko), *(const v8bf*)(arow + ko + 16));
      const bf16_t* wb = sh.w + kp * 512 + lane * 16;
      v16bf b = cat16(*(const v8bf*)wb, *(const v8bf*)(wb + 8));
      acc = __builtin_amdgcn_wmma_f32_16x16x32_bf16(false, a, false, b, (short)0, acc, false, false);
    }
    #pragma unroll
    for (int i = 0; i < 8; ++i) {
      float s  = fast_tanh(acc[i] + batt);
      int   m  = i + 8 * lhalf;
      float hv = bf2f(hs[((size_t)t * Bq + chunk * 16 + m) * Hq + ntile * 16 + lrow]);
      float mn = fmaxf(M[i], s);
      float sc = __expf(M[i] - mn);
      float p  = __expf(s - mn);
      D[i] = D[i] * sc + p;
      N[i] = N[i] * sc + p * hv;
      M[i] = mn;
    }
  }

  __syncthreads();   // done reading sh.w; reuse as merge buffers
  float* sM = sh.r;
  float* sD = sh.r + 16 * 256;
  float* sN = sh.r + 2 * 16 * 256;
  #pragma unroll
  for (int i = 0; i < 8; ++i) {
    int e = i * 32 + lane;
    sM[wave * 256 + e] = M[i];
    sD[wave * 256 + e] = D[i];
    sN[wave * 256 + e] = N[i];
  }
  __syncthreads();

  if (tid < 256) {
    int e = tid;
    float Ma = sM[e], Da = sD[e], Na = sN[e];
    for (int w = 1; w < 16; ++w) {
      float Mb = sM[w * 256 + e], Db = sD[w * 256 + e], Nb = sN[w * 256 + e];
      float Mn = fmaxf(Ma, Mb);
      float sa = __expf(Ma - Mn), sb = __expf(Mb - Mn);
      Da = Da * sa + Db * sb;
      Na = Na * sa + Nb * sb;
      Ma = Mn;
    }
    int i = e >> 5, l = e & 31;
    int m  = i + 8 * (l >> 4);
    int cl = l & 15;
    out[(size_t)(chunk * 16 + m) * Hq + ntile * 16 + cl] = Na / Da;
  }
}

// ---------------------------------------------------------------------------
extern "C" void kernel_launch(void* const* d_in, const int* in_sizes, int n_in,
                              void* d_out, int out_size, void* d_ws, size_t ws_size,
                              hipStream_t stream) {
  const float* x     = (const float*)d_in[0];
  const float* W_ih  = (const float*)d_in[1];
  const float* W_hh  = (const float*)d_in[2];
  const float* b_ih  = (const float*)d_in[3];
  const float* b_hh  = (const float*)d_in[4];
  const float* W_att = (const float*)d_in[5];
  const float* b_att = (const float*)d_in[6];
  float* out = (float*)d_out;

  char* ws = (char*)d_ws;
  bf16_t* Wpack = (bf16_t*)(ws);                          //  2,359,296 B
  bf16_t* Apack = (bf16_t*)(ws + 2359296);                //    524,288 B
  bf16_t* xpack = (bf16_t*)(ws + 2883584);                // 34,603,008 B
  bf16_t* hs    = (bf16_t*)(ws + 2883584 + 34603008);     // 67,108,864 B

  pack_gru_weights<<<(1179648 + 255) / 256, 256, 0, stream>>>(W_ih, W_hh, Wpack);
  pack_att_weights<<<(262144 + 255) / 256, 256, 0, stream>>>(W_att, Apack);
  pack_x<<<(Bq * Tq * Iq + 255) / 256, 256, 0, stream>>>(x, xpack);
  gru_recurrent<<<8, 1024, 0, stream>>>(xpack, b_ih, b_hh, Wpack, hs);
  attention_pool<<<dim3(32, 8), 512, 0, stream>>>(hs, Apack, b_att, out);
}
